// RGCN_73478300500627
// MI455X (gfx1250) — compile-verified, matrix operand order checked by reading, not run on hardware
//
#include <hip/hip_runtime.h>

typedef __bf16 bf16;
typedef __attribute__((ext_vector_type(16))) __bf16 v16bf;
typedef __attribute__((ext_vector_type(8)))  float  v8f;
typedef __attribute__((ext_vector_type(2)))  unsigned long long ull2;

union AFrag { v16bf v; ull2 u[2]; };

#define FEAT       128   // input feature width of each layer
#define KTOT       384   // 3 * 128 concatenated K (root + 2 relations)
#define NKT        12    // KTOT / 32
#define LDS_STRIDE 392   // 384 + 8 bf16 pad (row stride stays 16B-aligned: 784B)

// ---------------------------------------------------------------- utilities
__global__ void zero_f32(float* __restrict__ p, long long n) {
    long long i = (long long)blockIdx.x * blockDim.x + threadIdx.x;
    if (i < n) p[i] = 0.0f;
}

__global__ void convert_bf16(const float* __restrict__ src, bf16* __restrict__ dst, long long n) {
    long long i = (long long)blockIdx.x * blockDim.x + threadIdx.x;
    if (i < n) dst[i] = (bf16)src[i];
}

// Pack W = [Wroot(128xOUT) ; Wrel(2x128xOUT)] into WMMA B-fragment-major order:
// frag[((kt*CT + ct)*32 + lane)*16 + j]  holds B[k][col] with
//   col = ct*16 + (lane&15),  k = kt*32 + (lane>>4)*8 + (j<8 ? j : 8+j)
__global__ void pack_weights(const float* __restrict__ Wroot,
                             const float* __restrict__ Wrel,
                             bf16* __restrict__ frag, int OUT) {
    const int CT    = OUT / 16;
    const int total = NKT * CT * 32 * 16;
    int idx = blockIdx.x * blockDim.x + threadIdx.x;
    if (idx >= total) return;
    const int j    = idx & 15;
    const int lane = (idx >> 4) & 31;
    const int ct   = (idx >> 9) % CT;
    const int kt   = (idx >> 9) / CT;
    const int col  = ct * 16 + (lane & 15);
    const int k    = kt * 32 + (lane >> 4) * 8 + ((j < 8) ? j : (8 + j));
    const float v  = (k < 128) ? Wroot[k * OUT + col] : Wrel[(k - 128) * OUT + col];
    frag[idx] = (bf16)v;
}

// ------------------------------------------------------- graph aggregation
__global__ void count_edges(const long long* __restrict__ ei,
                            const long long* __restrict__ et,
                            float* __restrict__ cnt, long long E) {
    long long e = (long long)blockIdx.x * blockDim.x + threadIdx.x;
    if (e >= E) return;
    const long long d = ei[E + e];
    const long long t = et[e];
    atomicAdd(&cnt[d * 2 + t], 1.0f);
}

__global__ void recip_cnt(float* __restrict__ cnt, long long n) {
    long long i = (long long)blockIdx.x * blockDim.x + threadIdx.x;
    if (i < n) cnt[i] = 1.0f / fmaxf(cnt[i], 1.0f);
}

// 2 edges per 256-thread block; 128 lanes move one edge's 128 channels.
// agg (102MB) + feature table (25.6MB) both live in the 192MB L2.
__global__ void scatter_feat(const bf16* __restrict__ feat,
                             const long long* __restrict__ ei,
                             const long long* __restrict__ et,
                             float* __restrict__ agg, long long E) {
    const int ch = threadIdx.x & 127;
    const long long e = (long long)blockIdx.x * 2 + (threadIdx.x >> 7);
    if (e >= E) return;
    const long long s = ei[e];
    const long long d = ei[E + e];
    const long long t = et[e];
    const float v = (float)feat[s * FEAT + ch];
    atomicAdd(&agg[(d * 2 + t) * FEAT + ch], v);
}

// ------------------------------------------------------- WMMA transform
// out[n,:] = act( [feat(n) | agg(n,0)*inv | agg(n,1)*inv] @ Wfrag + bias )
template <int OUT, bool RELU, bool STORE_BF16>
__global__ void rgcn_transform(const bf16*  __restrict__ feat,   // [N,128] bf16
                               const float* __restrict__ agg,    // [N,2,128] f32
                               const float* __restrict__ inv,    // [N,2] f32
                               const bf16*  __restrict__ Bfrag,  // fragment-major weights
                               const float* __restrict__ bias,   // [OUT]
                               bf16*  __restrict__ out_bf,
                               float* __restrict__ out_f,
                               int N) {
    constexpr int CT  = OUT / 16;   // output col tiles
    constexpr int CPW = CT / 4;     // col tiles per wave (4 waves/block)
    __shared__ bf16 ldsA[16 * LDS_STRIDE];

    const int tid     = threadIdx.x;       // 0..127
    const int lane    = tid & 31;
    const int wave    = tid >> 5;          // 0..3
    const int rowbase = blockIdx.x * 16;

    // Stage A tile: 16 rows x 384 bf16 (x | scaled agg r=0 | scaled agg r=1)
    for (int idx = tid; idx < 16 * KTOT; idx += 128) {
        const int row = idx / KTOT;
        const int k   = idx - row * KTOT;
        const int n   = rowbase + row;
        bf16 v = (bf16)0.0f;
        if (n < N) {
            if (k < 128) {
                v = feat[(size_t)n * FEAT + k];
            } else {
                const int r  = (k - 128) >> 7;
                const int kk = (k - 128) & 127;
                v = (bf16)(agg[((size_t)n * 2 + r) * FEAT + kk] * inv[n * 2 + r]);
            }
        }
        ldsA[row * LDS_STRIDE + k] = v;
    }
    __syncthreads();

    const int mrow = lane & 15;
    const int sel  = lane >> 4;

    const v8f vzero = {0.f, 0.f, 0.f, 0.f, 0.f, 0.f, 0.f, 0.f};
    v8f acc[CPW];
#pragma unroll
    for (int j = 0; j < CPW; ++j) acc[j] = vzero;

#pragma unroll
    for (int kt = 0; kt < NKT; ++kt) {
        // A fragment from LDS (ISA 16-bit 16x32 layout, two 16B chunks / lane)
        AFrag fa;
        const bf16* ap = &ldsA[mrow * LDS_STRIDE + kt * 32 + sel * 8];
        fa.u[0] = *(const ull2*)(ap);
        fa.u[1] = *(const ull2*)(ap + 16);
#pragma unroll
        for (int j = 0; j < CPW; ++j) {
            const int ct = wave + j * 4;
            const v16bf b = *(const v16bf*)(Bfrag + (((size_t)kt * CT + ct) * 32 + lane) * 16);
            acc[j] = __builtin_amdgcn_wmma_f32_16x16x32_bf16(
                false, fa.v, false, b, (short)0, acc[j], false, false);
        }
    }

    // Store: C/D layout -> VGPR r, lanes 0-15: M=r ; lanes 16-31: M=8+r ; N=lane&15
#pragma unroll
    for (int j = 0; j < CPW; ++j) {
        const int ct  = wave + j * 4;
        const int col = ct * 16 + mrow;
#pragma unroll
        for (int r = 0; r < 8; ++r) {
            const int n = rowbase + r + sel * 8;
            if (n < N) {
                float val = acc[j][r] + bias[col];
                if (RELU) val = fmaxf(val, 0.0f);
                if (STORE_BF16) out_bf[(size_t)n * OUT + col] = (bf16)val;
                else            out_f [(size_t)n * OUT + col] = val;
            }
        }
    }
}

// ------------------------------------------------------- row-wise L2 norm (OUT=64)
__global__ void l2norm64(float* __restrict__ out, int N) {
    const int row  = blockIdx.x * (blockDim.x >> 5) + (threadIdx.x >> 5);
    const int lane = threadIdx.x & 31;
    if (row >= N) return;
    float2 v = *(float2*)(out + (size_t)row * 64 + lane * 2);
    float ss = v.x * v.x + v.y * v.y;
#pragma unroll
    for (int m = 16; m >= 1; m >>= 1) ss += __shfl_xor(ss, m, 32);
    const float scale = 1.0f / fmaxf(sqrtf(ss), 1e-12f);
    v.x *= scale; v.y *= scale;
    *(float2*)(out + (size_t)row * 64 + lane * 2) = v;
}

// ---------------------------------------------------------------- launcher
extern "C" void kernel_launch(void* const* d_in, const int* in_sizes, int n_in,
                              void* d_out, int out_size, void* d_ws, size_t ws_size,
                              hipStream_t stream) {
    const float*     x       = (const float*)d_in[0];
    const long long* ei      = (const long long*)d_in[1];
    const long long* et      = (const long long*)d_in[2];
    const float*     W1_rel  = (const float*)d_in[3];
    const float*     W1_root = (const float*)d_in[4];
    const float*     b1      = (const float*)d_in[5];
    const float*     W2_rel  = (const float*)d_in[6];
    const float*     W2_root = (const float*)d_in[7];
    const float*     b2      = (const float*)d_in[8];

    const long long N = in_sizes[0] / FEAT;
    const long long E = in_sizes[1] / 2;

    // workspace carve-out (256B aligned)
    char* w = (char*)d_ws;
    auto carve = [&](size_t bytes) {
        char* p = w;
        w += (bytes + 255) & ~(size_t)255;
        return p;
    };
    float* inv   = (float*)carve((size_t)N * 2 * sizeof(float));          // degree -> reciprocal
    bf16*  x_bf  = (bf16*) carve((size_t)N * FEAT * sizeof(bf16));
    bf16*  h_bf  = (bf16*) carve((size_t)N * FEAT * sizeof(bf16));
    float* agg   = (float*)carve((size_t)N * 2 * FEAT * sizeof(float));   // reused across layers
    bf16*  frag1 = (bf16*) carve((size_t)NKT * 8 * 32 * 16 * sizeof(bf16));
    bf16*  frag2 = (bf16*) carve((size_t)NKT * 4 * 32 * 16 * sizeof(bf16));
    float* outf  = (float*)d_out;

    const int B = 256;
    auto blocks = [&](long long n) { return (unsigned)((n + B - 1) / B); };

    // prep
    zero_f32<<<blocks(N * 2), B, 0, stream>>>(inv, N * 2);
    zero_f32<<<blocks(N * 2 * FEAT), B, 0, stream>>>(agg, N * 2 * FEAT);
    convert_bf16<<<blocks(N * FEAT), B, 0, stream>>>(x, x_bf, N * FEAT);
    pack_weights<<<blocks(NKT * 8 * 32 * 16), B, 0, stream>>>(W1_root, W1_rel, frag1, 128);
    pack_weights<<<blocks(NKT * 4 * 32 * 16), B, 0, stream>>>(W2_root, W2_rel, frag2, 64);

    // per-(dst, relation) mean denominators (identical for both layers)
    count_edges<<<blocks(E), B, 0, stream>>>(ei, et, inv, E);
    recip_cnt<<<blocks(N * 2), B, 0, stream>>>(inv, N * 2);

    const unsigned gridT = (unsigned)((N + 15) / 16);

    // layer 1
    scatter_feat<<<(unsigned)((E + 1) / 2), B, 0, stream>>>(x_bf, ei, et, agg, E);
    rgcn_transform<128, true, true><<<gridT, 128, 0, stream>>>(
        x_bf, agg, inv, frag1, b1, h_bf, nullptr, (int)N);

    // layer 2
    zero_f32<<<blocks(N * 2 * FEAT), B, 0, stream>>>(agg, N * 2 * FEAT);
    scatter_feat<<<(unsigned)((E + 1) / 2), B, 0, stream>>>(h_bf, ei, et, agg, E);
    rgcn_transform<64, false, false><<<gridT, 128, 0, stream>>>(
        h_bf, agg, inv, frag2, b2, nullptr, outf, (int)N);

    // final L2 normalize
    l2norm64<<<(unsigned)((N + 7) / 8), B, 0, stream>>>(outf, (int)N);

    (void)n_in; (void)out_size; (void)ws_size;
}
